// RingAttention_8512625180677
// MI455X (gfx1250) — matrix operand order repaired and checked by reading, not compiled, not run on hardware
//
#include <hip/hip_runtime.h>
#include <stdint.h>

// ---------------------------------------------------------------------------
// Causal MHA on gfx1250: convert/transpose -> qkv GEMM -> flash attention ->
// out GEMM. All matmuls on v_wmma_f32_16x16x32_bf16 (fp32 accum), all LDS
// staging through global_load_async_to_lds_b128 with double buffering.
// ---------------------------------------------------------------------------

typedef __attribute__((ext_vector_type(16))) __bf16 v16bf;
typedef __attribute__((ext_vector_type(8)))  float  v8f;

union Frag16 {
    uint4 q[2];   // 32 bytes
    v16bf v;      // 16 x bf16
};

__device__ __forceinline__ unsigned short f2bf(float f) {
    union { float f; unsigned int u; } c; c.f = f;
    unsigned int u = c.u;
    u += 0x7FFFu + ((u >> 16) & 1u);          // round-to-nearest-even
    return (unsigned short)(u >> 16);
}

// Async global->LDS copy, 16 bytes per lane.  LDS address = low 32 bits of
// the flat shared pointer.
__device__ __forceinline__ void async_cp_b128(void* lds_dst, const void* gsrc) {
    unsigned lds = (unsigned)(uintptr_t)lds_dst;
    unsigned long long ga = (unsigned long long)(uintptr_t)gsrc;
    asm volatile("global_load_async_to_lds_b128 %0, %1, off"
                 :: "v"(lds), "v"(ga) : "memory");
}
#define WAIT_ASYNC(n) asm volatile("s_wait_asynccnt %0" :: "i"(n) : "memory")

// ======================= fp32 -> bf16 elementwise ==========================
__global__ __launch_bounds__(256) void cvt_bf16_kernel(
    const float* __restrict__ src, unsigned short* __restrict__ dst, int n)
{
    int i = (blockIdx.x * 256 + threadIdx.x) * 8;
    if (i >= n) return;
    float4 a = *(const float4*)(src + i);
    float4 b = *(const float4*)(src + i + 4);
    union { unsigned short s[8]; uint4 q; } u;
    u.s[0]=f2bf(a.x); u.s[1]=f2bf(a.y); u.s[2]=f2bf(a.z); u.s[3]=f2bf(a.w);
    u.s[4]=f2bf(b.x); u.s[5]=f2bf(b.y); u.s[6]=f2bf(b.z); u.s[7]=f2bf(b.w);
    *(uint4*)(dst + i) = u.q;
}

// ================ fp32 [R][C] -> bf16 [C][R] (convert+transpose) ===========
__global__ __launch_bounds__(256) void cvt_transpose_kernel(
    const float* __restrict__ src, unsigned short* __restrict__ dst,
    int R, int C)
{
    __shared__ unsigned short Ts[64][72];
    const int r0 = blockIdx.y * 64, c0 = blockIdx.x * 64;
    const int t = threadIdx.x;
    for (int j = 0; j < 16; j++) {
        int idx = j * 256 + t;
        int r = idx >> 6, c = idx & 63;
        Ts[r][c] = f2bf(src[(size_t)(r0 + r) * C + c0 + c]);
    }
    __syncthreads();
    for (int j = 0; j < 16; j++) {
        int idx = j * 256 + t;
        int c = idx >> 6, r = idx & 63;
        dst[(size_t)(c0 + c) * R + r0 + r] = Ts[r][c];
    }
}

// ============================ bf16 GEMM ====================================
// A:[M][K] bf16 row-major, Bt:[N][K] bf16 row-major (pre-transposed weight).
// MODE 0: qkv scatter store (q scaled, v transposed), MODE 1: f32 store.
template<int MODE>
__global__ __launch_bounds__(256) void gemm_bf16_kernel(
    const unsigned short* __restrict__ A, const unsigned short* __restrict__ Bt,
    int K,
    unsigned short* __restrict__ qb, unsigned short* __restrict__ kb,
    unsigned short* __restrict__ vTb, float* __restrict__ C, int Nn)
{
    __shared__ __align__(16) unsigned short Als[2][128][32];
    __shared__ __align__(16) unsigned short Bls[2][128][32];

    const int t    = threadIdx.x;
    const int lane = t & 31;
    const int wave = t >> 5;
    const int half = (lane >> 4) & 1;
    const int ln   = lane & 15;
    const int m0   = blockIdx.y * 128;
    const int n0   = blockIdx.x * 128;
    const int wm   = (wave >> 1) * 32;
    const int wn   = (wave & 1) * 64;

    auto issue = [&](int k0, int buf) {
        for (int j = 0; j < 2; j++) {
            int idx = t * 2 + j;                 // 0..511
            int r = idx >> 2, seg = (idx & 3) * 8;
            async_cp_b128(&Als[buf][r][seg], A  + (size_t)(m0 + r) * K + k0 + seg);
            async_cp_b128(&Bls[buf][r][seg], Bt + (size_t)(n0 + r) * K + k0 + seg);
        }
    };

    v8f zero = {};
    v8f acc[2][4];
    for (int i = 0; i < 2; i++) for (int j = 0; j < 4; j++) acc[i][j] = zero;

    issue(0, 0);
    const int steps = K >> 5;                    // K/32
    for (int i = 0; i < steps; i++) {
        const int buf = i & 1;
        if (i + 1 < steps) { issue((i + 1) * 32, buf ^ 1); WAIT_ASYNC(4); }
        else               { WAIT_ASYNC(0); }
        __syncthreads();                         // tile i resident everywhere

        Frag16 a[2], b[4];
        for (int mr = 0; mr < 2; mr++) {
            const unsigned short* p = &Als[buf][wm + mr * 16 + ln][half * 8];
            a[mr].q[0] = *(const uint4*)p;
            a[mr].q[1] = *(const uint4*)(p + 16);
        }
        for (int nc = 0; nc < 4; nc++) {
            const unsigned short* p = &Bls[buf][wn + nc * 16 + ln][half * 8];
            b[nc].q[0] = *(const uint4*)p;
            b[nc].q[1] = *(const uint4*)(p + 16);
        }
        for (int mr = 0; mr < 2; mr++)
            for (int nc = 0; nc < 4; nc++)
                acc[mr][nc] = __builtin_amdgcn_wmma_f32_16x16x32_bf16(
                    false, a[mr].v, false, b[nc].v, (short)0, acc[mr][nc], false, false);

        __syncthreads();                         // reads done before reuse
    }

    if (MODE == 0) {
        const float scale = 0.08838834764831845f; // 128^-0.5
        for (int mr = 0; mr < 2; mr++)
            for (int nc = 0; nc < 4; nc++)
                for (int r = 0; r < 8; r++) {
                    int m = m0 + wm + mr * 16 + half * 8 + r;
                    int n = n0 + wn + nc * 16 + ln;
                    float v = acc[mr][nc][r];
                    int which = n >> 11;          // 0:q 1:k 2:v
                    int inner = n & 2047;
                    int h  = inner >> 7, dh = inner & 127;
                    int bb = m >> 11,    nn = m & 2047;
                    if (which == 0) {
                        qb[((size_t)((bb * 16 + h) * 2048 + nn)) * 128 + dh] = f2bf(v * scale);
                    } else if (which == 1) {
                        kb[((size_t)((bb * 16 + h) * 2048 + nn)) * 128 + dh] = f2bf(v);
                    } else {                      // V stored transposed: [b,h,dh,n]
                        vTb[((size_t)((bb * 16 + h) * 128 + dh)) * 2048 + nn] = f2bf(v);
                    }
                }
    } else {
        for (int mr = 0; mr < 2; mr++)
            for (int nc = 0; nc < 4; nc++)
                for (int r = 0; r < 8; r++) {
                    int m = m0 + wm + mr * 16 + half * 8 + r;
                    int n = n0 + wn + nc * 16 + ln;
                    C[(size_t)m * Nn + n] = acc[mr][nc][r];
                }
    }
}

// =================== Flash attention (causal, online softmax) ==============
// One block = 4 waves = 64 query rows of one (b,h).
// qb/kb: [B,H,N,Dh] bf16.  vTb: [B,H,Dh,N] bf16 (transposed V).
__global__ __launch_bounds__(128) void attn_kernel(
    const unsigned short* __restrict__ qb, const unsigned short* __restrict__ kb,
    const unsigned short* __restrict__ vTb, unsigned short* __restrict__ ob)
{
    const int Nseq = 2048, Dh = 128;
    __shared__ __align__(16) unsigned short Qs[64][128];       // 16KB
    __shared__ __align__(16) unsigned short Ks[2][64][128];    // 32KB
    __shared__ __align__(16) unsigned short Vt[2][128][64];    // 32KB [dh][key]
    __shared__ __align__(16) unsigned short Ps[4][16][64];     // 8KB per-wave P

    const int t    = threadIdx.x;
    const int lane = t & 31;
    const int w    = t >> 5;
    const int half = (lane >> 4) & 1;
    const int ln   = lane & 15;
    const int qt   = blockIdx.x & 31;     // 32 query tiles of 64
    const int bh   = blockIdx.x >> 5;     // b*16 + h
    const int q0   = qt * 64;
    const size_t base  = (size_t)bh * Nseq * Dh;   // for q/k
    const size_t baseT = (size_t)bh * Dh * Nseq;   // for vT

    auto issueKV = [&](int kt, int buf) {
        const int j0 = kt * 64;
        for (int j = 0; j < 8; j++) {             // K tile: 64 x 128
            int idx = t * 8 + j;
            int r = idx >> 4, c = (idx & 15) * 8;
            async_cp_b128(&Ks[buf][r][c], kb + base + (size_t)(j0 + r) * Dh + c);
        }
        for (int j = 0; j < 8; j++) {             // V^T tile: 128 x 64
            int idx = t * 8 + j;
            int d = idx >> 3, c = (idx & 7) * 8;
            async_cp_b128(&Vt[buf][d][c], vTb + baseT + (size_t)d * Nseq + j0 + c);
        }
    };

    issueKV(0, 0);                                // prefetch first K/V tile

    for (int j = 0; j < 8; j++) {                 // Q tile (sync copy)
        int idx = t * 8 + j;
        int r = idx >> 4, c = (idx & 15) * 8;
        *(uint4*)&Qs[r][c] = *(const uint4*)(qb + base + (size_t)(q0 + r) * Dh + c);
    }
    __syncthreads();

    Frag16 aq[4];
    for (int kc = 0; kc < 4; kc++) {
        const unsigned short* p = &Qs[w * 16 + ln][kc * 32 + half * 8];
        aq[kc].q[0] = *(const uint4*)p;
        aq[kc].q[1] = *(const uint4*)(p + 16);
    }

    float mrow[8], lrow[8];
    v8f zero = {};
    v8f o[8];
    for (int i = 0; i < 8; i++) { o[i] = zero; mrow[i] = -3.0e38f; lrow[i] = 0.f; }

    for (int kt = 0; kt <= qt; kt++) {
        const int buf = kt & 1;
        const int j0  = kt * 64;
        if (kt < qt) { issueKV(kt + 1, buf ^ 1); WAIT_ASYNC(16); }
        else         { WAIT_ASYNC(0); }
        __syncthreads();                          // tile kt resident

        // S = Q * K^T  (16x64 per wave)
        v8f s[4];
        for (int nc = 0; nc < 4; nc++) s[nc] = zero;
        for (int nc = 0; nc < 4; nc++)
            for (int kc = 0; kc < 4; kc++) {
                Frag16 bk;
                const unsigned short* p = &Ks[buf][nc * 16 + ln][kc * 32 + half * 8];
                bk.q[0] = *(const uint4*)p;
                bk.q[1] = *(const uint4*)(p + 16);
                s[nc] = __builtin_amdgcn_wmma_f32_16x16x32_bf16(
                    false, aq[kc].v, false, bk.v, (short)0, s[nc], false, false);
            }

        if (kt == qt) {                           // diagonal tile: causal mask
            for (int nc = 0; nc < 4; nc++)
                for (int r = 0; r < 8; r++) {
                    int qi = q0 + w * 16 + half * 8 + r;
                    int kj = j0 + nc * 16 + ln;
                    if (kj > qi) s[nc][r] = -3.0e38f;
                }
        }

        // online softmax (16-lane half owns 8 rows of the wave's 16)
        float pvv[4][8];
        for (int r = 0; r < 8; r++) {
            float mx = s[0][r];
            for (int nc = 1; nc < 4; nc++) mx = fmaxf(mx, s[nc][r]);
            for (int off = 1; off < 16; off <<= 1) mx = fmaxf(mx, __shfl_xor(mx, off, 32));
            float mn = fmaxf(mrow[r], mx);
            float alpha = __expf(mrow[r] - mn);
            float rs = 0.f;
            for (int nc = 0; nc < 4; nc++) { float e = __expf(s[nc][r] - mn); pvv[nc][r] = e; rs += e; }
            for (int off = 1; off < 16; off <<= 1) rs += __shfl_xor(rs, off, 32);
            lrow[r] = lrow[r] * alpha + rs;
            mrow[r] = mn;
            for (int dc = 0; dc < 8; dc++) o[dc][r] *= alpha;
        }

        // P (C-layout) -> per-wave LDS; same-wave LDS is in-order, no barrier
        for (int nc = 0; nc < 4; nc++)
            for (int r = 0; r < 8; r++)
                Ps[w][half * 8 + r][nc * 16 + ln] = f2bf(pvv[nc][r]);

        // O += P @ V
        for (int kc2 = 0; kc2 < 2; kc2++) {
            Frag16 ap;
            const unsigned short* pp = &Ps[w][ln][kc2 * 32 + half * 8];
            ap.q[0] = *(const uint4*)pp;
            ap.q[1] = *(const uint4*)(pp + 16);
            for (int dc = 0; dc < 8; dc++) {
                Frag16 bv;
                const unsigned short* vp = &Vt[buf][dc * 16 + ln][kc2 * 32 + half * 8];
                bv.q[0] = *(const uint4*)vp;
                bv.q[1] = *(const uint4*)(vp + 16);
                o[dc] = __builtin_amdgcn_wmma_f32_16x16x32_bf16(
                    false, ap.v, false, bv.v, (short)0, o[dc], false, false);
            }
        }
        __syncthreads();                          // K/V reads done before reuse
    }

    // normalize, store bf16 attn output, layout [B, N, H*Dh] = [4096,2048]
    const int bq = bh >> 4;
    const int hh = bh & 15;
    for (int dc = 0; dc < 8; dc++)
        for (int r = 0; r < 8; r++) {
            int row = q0 + w * 16 + half * 8 + r;
            float vout = o[dc][r] / lrow[r];
            size_t dst = ((size_t)(bq * 2048 + row)) * 2048 + hh * 128 + dc * 16 + ln;
            ob[dst] = f2bf(vout);
        }
}

// ============================== launcher ===================================
extern "C" void kernel_launch(void* const* d_in, const int* in_sizes, int n_in,
                              void* d_out, int out_size, void* d_ws, size_t ws_size,
                              hipStream_t stream) {
    const float* x     = (const float*)d_in[0];
    const float* w_qkv = (const float*)d_in[1];
    const float* w_out = (const float*)d_in[2];
    float* out = (float*)d_out;

    const size_t nx  = (size_t)4096 * 2048;    // x / per-tensor q,k,v / ob
    const size_t nwq = (size_t)2048 * 6144;
    const size_t nwo = (size_t)2048 * 2048;

    unsigned short* xb    = (unsigned short*)d_ws;
    unsigned short* wqkvT = xb    + nx;        // [6144][2048]
    unsigned short* woutT = wqkvT + nwq;       // [2048][2048]
    unsigned short* qb    = woutT + nwo;       // [B,H,N,Dh]
    unsigned short* kb    = qb + nx;
    unsigned short* vTb   = kb + nx;           // [B,H,Dh,N]
    unsigned short* ob    = vTb + nx;          // [B,N,H*Dh]

    cvt_bf16_kernel<<<dim3((unsigned)(nx / (256 * 8))), 256, 0, stream>>>(x, xb, (int)nx);
    cvt_transpose_kernel<<<dim3(6144 / 64, 2048 / 64), 256, 0, stream>>>(w_qkv, wqkvT, 2048, 6144);
    cvt_transpose_kernel<<<dim3(2048 / 64, 2048 / 64), 256, 0, stream>>>(w_out, woutT, 2048, 2048);

    dim3 g1(6144 / 128, 4096 / 128);
    gemm_bf16_kernel<0><<<g1, 256, 0, stream>>>(xb, wqkvT, 2048, qb, kb, vTb, nullptr, 6144);

    attn_kernel<<<dim3(32 * 32), 128, 0, stream>>>(qb, kb, vTb, ob);

    dim3 g2(2048 / 128, 4096 / 128);
    gemm_bf16_kernel<1><<<g2, 256, 0, stream>>>(ob, woutT, 2048, nullptr, nullptr, nullptr, out, 2048);
}
